// EntropyDiffRegNet_lite_9929964388406
// MI455X (gfx1250) — matrix operand order repaired and checked by reading, not compiled, not run on hardware
//
#include <hip/hip_runtime.h>

typedef _Float16 half_t;
typedef __attribute__((ext_vector_type(16))) _Float16 v16h;   // 32 B
typedef __attribute__((ext_vector_type(8)))  _Float16 v8h;    // 16 B
typedef __attribute__((ext_vector_type(8)))  float    v8f;

#define DEVI __device__ __forceinline__

namespace {
constexpr int   Bn = 4, Cch = 64, OCh = 32, Hd = 256, Wd = 256;
constexpr long  Np = (long)Hd * Wd;              // 65536 pixels per image
constexpr float EPSc = 1e-8f;
constexpr float SIGMAc = 0.08f, TAUc = 0.6f;
constexpr float W_FLOORc = 0.15f, W_BETAc = 0.6f, FLOW_ALPHAc = 0.25f;
constexpr float BIGF = 3.402823e38f;
}

DEVI int imaxi(int a, int b) { return a > b ? a : b; }
DEVI int imini(int a, int b) { return a < b ? a : b; }
DEVI int refl(int i, int n) { i = i < 0 ? -i : i; i = i >= n ? 2 * n - 2 - i : i; return i; }

DEVI void atomicMaxF(float* a, float v) {
  unsigned int* p = (unsigned int*)a; unsigned int old = *p;
  while (__uint_as_float(old) < v) {
    unsigned int assumed = old;
    old = atomicCAS(p, assumed, __float_as_uint(v));
    if (old == assumed) break;
  }
}
DEVI void atomicMinF(float* a, float v) {
  unsigned int* p = (unsigned int*)a; unsigned int old = *p;
  while (__uint_as_float(old) > v) {
    unsigned int assumed = old;
    old = atomicCAS(p, assumed, __float_as_uint(v));
    if (old == assumed) break;
  }
}

// WMMA 16x16x32 f16 A-operand K map: element i of v16h -> K index (k = hi*16 + i, matches B).
DEVI int akmap(int i, int hi) {
  return hi * 16 + i;
}
// B-operand: k = hi*16 + i  (contiguous per lane -> single v16h load in
// channels-last layout).

// ---------------------------------------------------------------------------
// Scratch init: image/entropy min-max atomic slots only
// ---------------------------------------------------------------------------
__global__ void k_init(float* s) {
  int i = blockIdx.x * blockDim.x + threadIdx.x;
  if (i >= 24) return;
  bool isMin = (i < 4) || (i >= 8 && i < 12) || (i >= 16 && i < 20);
  s[i] = isMin ? BIGF : -BIGF;
}

// ---------------------------------------------------------------------------
// Weight pre-pack into per-lane WMMA A layout.
// packed[((coT*KP + kp)*(CIN/32) + cbi)*32 + lane][16]  <- w[co][ci][kp]
// ---------------------------------------------------------------------------
__global__ void k_pack(const float* w, half_t* packed, int CO, int CIN, int KP) {
  int t = blockIdx.x * blockDim.x + threadIdx.x;
  int total = (CO / 16) * KP * (CIN / 32) * 32;
  if (t >= total) return;
  int lane = t & 31, rest = t >> 5;
  int nC = CIN / 32;
  int cbi = rest % nC; rest /= nC;
  int kp = rest % KP;  int coT = rest / KP;
  int nl = lane & 15, hi = lane >> 4;
  half_t* dst = packed + (size_t)t * 16;
#pragma unroll
  for (int i = 0; i < 16; i++) {
    int ci = cbi * 32 + akmap(i, hi);
    dst[i] = (half_t)w[((coT * 16 + nl) * CIN + ci) * KP + kp];
  }
}

// ---------------------------------------------------------------------------
// Per-batch min/max of X (or sc*(X+Y)) over spatial dim, via LDS + atomics
// ---------------------------------------------------------------------------
__global__ void k_minmax2(const float* X, const float* Y, float sc,
                          float* rmin, float* rmax) {
  int b = blockIdx.y;
  __shared__ float smin[256], smax[256];
  float mn = BIGF, mx = -BIGF;
  for (long p = blockIdx.x * blockDim.x + threadIdx.x; p < Np;
       p += (long)gridDim.x * blockDim.x) {
    float v = X[(long)b * Np + p];
    if (Y) v = sc * (v + Y[(long)b * Np + p]);
    mn = fminf(mn, v); mx = fmaxf(mx, v);
  }
  smin[threadIdx.x] = mn; smax[threadIdx.x] = mx;
  __syncthreads();
  for (int s = 128; s; s >>= 1) {
    if (threadIdx.x < s) {
      smin[threadIdx.x] = fminf(smin[threadIdx.x], smin[threadIdx.x + s]);
      smax[threadIdx.x] = fmaxf(smax[threadIdx.x], smax[threadIdx.x + s]);
    }
    __syncthreads();
  }
  if (threadIdx.x == 0) { atomicMinF(&rmin[b], smin[0]); atomicMaxF(&rmax[b], smax[0]); }
}

// ---------------------------------------------------------------------------
// Soft-histogram bin weights
// ---------------------------------------------------------------------------
__global__ void k_bins(const float* img, const float* rmin, const float* rmax,
                       float* bins) {
  long i = (long)blockIdx.x * blockDim.x + threadIdx.x;   // Bn*Np
  if (i >= Bn * Np) return;
  int b = (int)(i >> 16); long p = i & 65535;
  float mn = rmin[b], mx = rmax[b];
  float x = (img[i] - mn) / (mx - mn + EPSc);
  x = fminf(fmaxf(x, 0.f), 1.f);
  float w[16]; float s = 0.f;
#pragma unroll
  for (int k = 0; k < 16; k++) {
    float d = (x - (float)k * (1.f / 15.f)) / (SIGMAc + EPSc);
    w[k] = __expf(-0.5f * d * d);
    s += w[k];
  }
  float inv = 1.f / (s + EPSc);
#pragma unroll
  for (int k = 0; k < 16; k++) bins[((long)(b * 16 + k)) * Np + p] = w[k] * inv;
}

// Separable 9-tap reflect sum-pool, horizontal pass
__global__ void k_poolh(const float* in, float* out) {
  long i = (long)blockIdx.x * blockDim.x + threadIdx.x;   // Bn*16*Np
  if (i >= (long)Bn * 16 * Np) return;
  long bk = i >> 16; int p = (int)(i & 65535);
  int y = p >> 8, x = p & 255;
  float s = 0.f;
#pragma unroll
  for (int d = -4; d <= 4; d++) s += in[bk * Np + (y << 8) + refl(x + d, Wd)];
  out[i] = s;
}

// Vertical pass fused with Shannon entropy over the 16 bins
__global__ void k_poolv_ent(const float* in, float* Hout) {
  long i = (long)blockIdx.x * blockDim.x + threadIdx.x;   // Bn*Np
  if (i >= Bn * Np) return;
  int b = (int)(i >> 16); int p = (int)(i & 65535);
  int y = p >> 8, x = p & 255;
  float ent = 0.f;
#pragma unroll
  for (int k = 0; k < 16; k++) {
    float s = 0.f;
#pragma unroll
    for (int d = -4; d <= 4; d++)
      s += in[((long)(b * 16 + k)) * Np + (refl(y + d, Hd) << 8) + x];
    float pr = fmaxf(s * (1.f / 81.f), EPSc);
    ent -= pr * __log2f(pr);
  }
  Hout[i] = ent;
}

__global__ void k_weight(const float* H1, const float* H2,
                         const float* gmin, const float* gmax, float* Wpre) {
  long i = (long)blockIdx.x * blockDim.x + threadIdx.x;   // Bn*Np
  if (i >= Bn * Np) return;
  int b = (int)(i >> 16);
  float h1 = H1[i], h2 = H2[i];
  float dH = fabsf(h1 - h2);
  float w = __expf(-dH / (TAUc + EPSc));
  w = fminf(fmaxf(w, 0.f), 1.f);
  w = sqrtf(w);                                   // ** W_GAMMA (=0.5)
  float hs = 0.5f * (h1 + h2);
  float g = (hs - gmin[b]) / (gmax[b] - gmin[b] + EPSc);
  float Wv = fminf(fmaxf(w * g, 0.f), 1.f);
  Wv = Wv * (1.f - W_FLOORc) + W_FLOORc;
  Wpre[i] = Wv;
}

__global__ void k_smooth(const float* in, float* Ws) {
  long i = (long)blockIdx.x * blockDim.x + threadIdx.x;   // Bn*Np
  if (i >= Bn * Np) return;
  int b = (int)(i >> 16); int p = (int)(i & 65535);
  int y = p >> 8, x = p & 255;
  float s = 0.f;
#pragma unroll
  for (int dy = -1; dy <= 1; dy++)
#pragma unroll
    for (int dx = -1; dx <= 1; dx++)
      s += in[(long)b * Np + (refl(y + dy, Hd) << 8) + refl(x + dx, Wd)];
  Ws[i] = fminf(fmaxf(s * (1.f / 9.f), 0.f), 1.f);
}

// ---------------------------------------------------------------------------
// Planar f32 [b][64][p] -> channels-last f16 [b][p][64], LDS transpose tile.
// ---------------------------------------------------------------------------
__global__ __launch_bounds__(256) void k_tr(const float* in, half_t* out) {
  __shared__ half_t tile[64 * 32];
  int b = blockIdx.y;
  long p0 = (long)blockIdx.x * 32;
  int t = threadIdx.x;
#pragma unroll
  for (int j = 0; j < 8; j++) {
    int idx = t + j * 256;            // 2048 = 64*32
    int c = idx >> 5, px = idx & 31;
    tile[c * 32 + px] = (half_t)in[((long)b * 64 + c) * Np + p0 + px];
  }
  __syncthreads();
#pragma unroll
  for (int j = 0; j < 8; j++) {
    int idx = t + j * 256;
    int px = idx >> 6, c = idx & 63;
    out[((long)b * Np + p0 + px) * 64 + c] = tile[c * 32 + px];
  }
}

// ---------------------------------------------------------------------------
// 1x1 conv 128->64 + leaky_relu -> channels-last f16.
// ---------------------------------------------------------------------------
__global__ __launch_bounds__(256) void k_conv1x1_f2(const float* in, const half_t* wp,
                                                    const float* bias, half_t* out) {
  int wv = threadIdx.x >> 5, lane = threadIdx.x & 31;
  int hi = lane >> 4;
  int coT = wv & 3, pxT = wv >> 2;
  long gp0 = (long)blockIdx.x * 32;
  int b = (int)(gp0 >> 16); long p = (gp0 & 65535) + pxT * 16 + (lane & 15);
  int co0 = coT * 16;
  v8f c = {};
#pragma unroll
  for (int cbi = 0; cbi < 4; cbi++) {
    v16h a = *(const v16h*)&wp[(((size_t)coT * 4 + cbi) * 32 + lane) * 16];
    v16h bm;
#pragma unroll
    for (int i = 0; i < 16; i++) {
      int kb = hi * 16 + i;
      bm[i] = (half_t)in[((long)b * 128 + cbi * 32 + kb) * Np + p];
    }
    c = __builtin_amdgcn_wmma_f32_16x16x32_f16(false, a, false, bm, (short)0, c,
                                               false, false);
  }
  v8h h8;
#pragma unroll
  for (int r = 0; r < 8; r++) {
    float v = c[r] + bias[co0 + 8 * hi + r];
    h8[r] = (half_t)(v > 0.f ? v : 0.01f * v);   // leaky_relu
  }
  *(v8h*)&out[((long)b * Np + p) * 64 + co0 + 8 * hi] = h8;
}

// ---------------------------------------------------------------------------
// 3x3 conv 64->64 (+ReLU, optional f16 residual), channels-last in/out.
// LDS-staged halo tile, v16h B reads shared by all 4 co-tiles, packed A.
// ---------------------------------------------------------------------------
__global__ __launch_bounds__(256) void k_conv3x3(const half_t* in, const half_t* wp,
                                                 const float* bias, half_t* out_h,
                                                 const half_t* resid) {
  __shared__ half_t sh[3 * 34 * 64];              // 13056 B
  int wv = threadIdx.x >> 5, lane = threadIdx.x & 31;
  int nl = lane & 15, hi = lane >> 4;
  int coT = wv & 3, pxT = wv >> 2;
  int b = blockIdx.z, y = blockIdx.y;
  int x0 = blockIdx.x * 32;
  __builtin_prefetch(&in[(((long)b * Hd + y) * Wd + x0) * 64], 0, 1);

  // cooperative halo fill: 816 uint4 slots (= 3*34*8), zero-filled borders
  for (int slot = threadIdx.x; slot < 816; slot += 256) {
    int cg = slot & 7;
    int pix = slot >> 3;
    int r = pix / 34, px = pix % 34;
    int yy = y + r - 1, xx = x0 + px - 1;
    bool ok = (yy >= 0) && (yy < Hd) && (xx >= 0) && (xx < Wd);
    int yyc = imini(imaxi(yy, 0), Hd - 1), xxc = imini(imaxi(xx, 0), Wd - 1);
    uint4 v = ok ? *(const uint4*)&in[(((long)b * Hd + yyc) * Wd + xxc) * 64 + cg * 8]
                 : make_uint4(0u, 0u, 0u, 0u);
    *(uint4*)&sh[(size_t)slot * 8] = v;
  }
  __syncthreads();

  int co0 = coT * 16;
  v8f c = {};
  for (int kp = 0; kp < 9; kp++) {
    int ky = kp / 3, kx = kp % 3;                 // 0..2 (tile-relative)
    int tx = pxT * 16 + nl + kx;                  // 0..33
#pragma unroll
    for (int cbi = 0; cbi < 2; cbi++) {
      v16h a = *(const v16h*)&wp[((((size_t)coT * 9 + kp) * 2 + cbi) * 32 + lane) * 16];
      v16h bm = *(const v16h*)&sh[((size_t)ky * 34 + tx) * 64 + cbi * 32 + hi * 16];
      c = __builtin_amdgcn_wmma_f32_16x16x32_f16(false, a, false, bm, (short)0, c,
                                                 false, false);
    }
  }
  long base = (((long)b * Hd + y) * Wd + x0 + pxT * 16 + nl) * 64 + co0 + 8 * hi;
  v8h h8;
  if (resid) {
    v8h rv = *(const v8h*)&resid[base];
#pragma unroll
    for (int r = 0; r < 8; r++)
      h8[r] = (half_t)(fmaxf(c[r] + bias[co0 + 8 * hi + r], 0.f) + (float)rv[r]);
  } else {
#pragma unroll
    for (int r = 0; r < 8; r++)
      h8[r] = (half_t)fmaxf(c[r] + bias[co0 + 8 * hi + r], 0.f);
  }
  *(v8h*)&out_h[base] = h8;
}

// ---------------------------------------------------------------------------
// Single-pass online softmax partials: each block emits per-channel
// (max, sum-exp) over its 1024-pixel chunk.  part arrays: [chunk][b*64+c].
// ---------------------------------------------------------------------------
__global__ void k_attpart(const half_t* A, const half_t* Bm, const float* Ws,
                          float* pmax, float* psum) {
  int b = blockIdx.y;
  int c = threadIdx.x & 63, ps = threadIdx.x >> 6;
  __shared__ float sm[256], ss[256];
  float mx = -BIGF, s = 0.f;
  long p0 = (long)blockIdx.x * 1024;
  for (int j = 0; j < 256; j++) {
    long p = p0 + ps + 4 * j;
    long idx = ((long)b * Np + p) * 64 + c;
    float l = (float)A[idx] * (float)Bm[idx] +
              W_BETAc * __logf(fmaxf(Ws[(long)b * Np + p], EPSc));
    float nm = fmaxf(mx, l);
    s = s * __expf(mx - nm) + __expf(l - nm);
    mx = nm;
  }
  sm[threadIdx.x] = mx; ss[threadIdx.x] = s;
  __syncthreads();
  if (threadIdx.x < 128) {
    float m1 = sm[threadIdx.x], m2 = sm[threadIdx.x + 128];
    float m = fmaxf(m1, m2);
    ss[threadIdx.x] = ss[threadIdx.x] * __expf(m1 - m) + ss[threadIdx.x + 128] * __expf(m2 - m);
    sm[threadIdx.x] = m;
  }
  __syncthreads();
  if (threadIdx.x < 64) {
    float m1 = sm[threadIdx.x], m2 = sm[threadIdx.x + 64];
    float m = fmaxf(m1, m2);
    float sv = ss[threadIdx.x] * __expf(m1 - m) + ss[threadIdx.x + 64] * __expf(m2 - m);
    int o = blockIdx.x * 256 + b * 64 + threadIdx.x;
    pmax[o] = m; psum[o] = sv;
  }
}

// Tree-merge the 64 per-chunk partials -> final (max, denom) per (b,c)
__global__ void k_attfinal(const float* pmax, const float* psum,
                           float* fmax, float* fsum) {
  int bc = threadIdx.x;                           // 256 = Bn*64
  float m = -BIGF, s = 0.f;
  for (int ch = 0; ch < 64; ch++) {
    float m2 = pmax[ch * 256 + bc], s2 = psum[ch * 256 + bc];
    float nm = fmaxf(m, m2);
    s = s * __expf(m - nm) + s2 * __expf(m2 - nm);
    m = nm;
  }
  fmax[bc] = m; fsum[bc] = s;
}

// Fused: f = A1*att1 + f1 + A2*att2 + f2   (all channels-last f16)
__global__ void k_combine(const half_t* A1, const half_t* B2, const half_t* A2,
                          const half_t* B1, const half_t* f1h, const half_t* f2h,
                          const float* Ws, const float* max1, const float* sum1,
                          const float* max2, const float* sum2, half_t* fh) {
  long i = (long)blockIdx.x * blockDim.x + threadIdx.x;   // Bn*Np*64
  if (i >= (long)Bn * Np * 64) return;
  int c = (int)(i & 63); long bp = i >> 6;
  long p = bp & 65535; int b = (int)(bp >> 16);
  int bc = b * 64 + c;
  float lb = W_BETAc * __logf(fmaxf(Ws[(long)b * Np + p], EPSc));
  float a1 = (float)A1[i], b2 = (float)B2[i];
  float a2 = (float)A2[i], b1 = (float)B1[i];
  float att1 = __expf(a1 * b2 + lb - max1[bc]) / sum1[bc];
  float att2 = __expf(a2 * b1 + lb - max2[bc]) / sum2[bc];
  float v = a1 * att1 + (float)f1h[i] + a2 * att2 + (float)f2h[i];
  fh[i] = (half_t)v;
}

// Flow head: 3x3 conv 64->2, tanh, entropy gate (channels-last input)
__global__ void k_flow(const half_t* f, const float* wfl, const float* bfl,
                       const float* Ws, float* flow) {
  long i = (long)blockIdx.x * blockDim.x + threadIdx.x;   // Bn*Np
  if (i >= Bn * Np) return;
  int b = (int)(i >> 16); int p = (int)(i & 65535);
  int y = p >> 8, x = p & 255;
  float a0 = 0.f, a1 = 0.f;
#pragma unroll
  for (int kp = 0; kp < 9; kp++) {
    int yy = y + kp / 3 - 1, xx = x + kp % 3 - 1;
    bool ok = (yy >= 0) && (yy < Hd) && (xx >= 0) && (xx < Wd);
    int yyc = imini(imaxi(yy, 0), Hd - 1), xxc = imini(imaxi(xx, 0), Wd - 1);
    long base = (((long)b * Hd + yyc) * Wd + xxc) * 64;
    for (int ci = 0; ci < 64; ci++) {
      float v = ok ? (float)f[base + ci] : 0.f;
      a0 += v * wfl[(0 * 64 + ci) * 9 + kp];
      a1 += v * wfl[(1 * 64 + ci) * 9 + kp];
    }
  }
  float gate = FLOW_ALPHAc + (1.f - FLOW_ALPHAc) * Ws[i];
  flow[((long)b * 2 + 0) * Np + p] = tanhf(a0 + bfl[0]) * gate;
  flow[((long)b * 2 + 1) * Np + p] = tanhf(a1 + bfl[1]) * gate;
}

// ---------------------------------------------------------------------------
// Fused bilinear 2x upsample (align_corners) + 1x1 conv 64->32 + leaky_relu.
// ---------------------------------------------------------------------------
__global__ __launch_bounds__(256) void k_upconv(const half_t* f, const half_t* wp,
                                                const float* bias, float* out) {
  int wv = threadIdx.x >> 5, lane = threadIdx.x & 31;
  int nl = lane & 15, hi = lane >> 4;
  int coT = wv & 1, pxT = wv >> 1;
  int b = blockIdx.z, Y = blockIdx.y;
  int X0 = blockIdx.x * 64 + pxT * 16;
  int X = X0 + nl;
  const float sc = 255.f / 511.f;
  float sy = (float)Y * sc; int y0 = (int)sy; int y1 = imini(y0 + 1, 255); float fy = sy - (float)y0;
  float sx = (float)X * sc; int x0 = (int)sx; int x1 = imini(x0 + 1, 255); float fx = sx - (float)x0;
  long r00 = ((long)b * Np + (y0 << 8) + x0) * 64;
  long r01 = ((long)b * Np + (y0 << 8) + x1) * 64;
  long r10 = ((long)b * Np + (y1 << 8) + x0) * 64;
  long r11 = ((long)b * Np + (y1 << 8) + x1) * 64;
  int co0 = coT * 16;
  v8f c = {};
#pragma unroll
  for (int cbi = 0; cbi < 2; cbi++) {
    int ko = cbi * 32 + hi * 16;
    v16h c00 = *(const v16h*)&f[r00 + ko];
    v16h c01 = *(const v16h*)&f[r01 + ko];
    v16h c10 = *(const v16h*)&f[r10 + ko];
    v16h c11 = *(const v16h*)&f[r11 + ko];
    v16h a = *(const v16h*)&wp[(((size_t)coT * 2 + cbi) * 32 + lane) * 16];
    v16h bm;
#pragma unroll
    for (int i = 0; i < 16; i++) {
      float top = (float)c00[i] * (1.f - fx) + (float)c01[i] * fx;
      float bot = (float)c10[i] * (1.f - fx) + (float)c11[i] * fx;
      bm[i] = (half_t)(top * (1.f - fy) + bot * fy);
    }
    c = __builtin_amdgcn_wmma_f32_16x16x32_f16(false, a, false, bm, (short)0, c,
                                               false, false);
  }
#pragma unroll
  for (int r = 0; r < 8; r++) {
    int co = co0 + 8 * hi + r;
    float v = c[r] + bias[co];
    v = v > 0.f ? v : 0.01f * v;                 // leaky_relu
    out[(((long)b * 32 + co) * 512 + Y) * 512 + X0 + nl] = v;
  }
}

// ---------------------------------------------------------------------------
extern "C" void kernel_launch(void* const* d_in, const int* in_sizes, int n_in,
                              void* d_out, int out_size, void* d_ws, size_t ws_size,
                              hipStream_t stream) {
  (void)in_sizes; (void)n_in; (void)out_size; (void)ws_size;
  const float* img1   = (const float*)d_in[0];
  const float* img2   = (const float*)d_in[1];
  const float* f1     = (const float*)d_in[2];
  const float* f2cat  = (const float*)d_in[3];
  const float* w_c11  = (const float*)d_in[4];
  const float* b_c11  = (const float*)d_in[5];
  const float* w_ca1  = (const float*)d_in[6];
  const float* b_ca1  = (const float*)d_in[7];
  const float* w_ca2  = (const float*)d_in[8];
  const float* b_ca2  = (const float*)d_in[9];
  const float* w_fo   = (const float*)d_in[10];
  const float* b_fo   = (const float*)d_in[11];
  const float* w_fl   = (const float*)d_in[12];
  const float* b_fl   = (const float*)d_in[13];
  const float* w_up   = (const float*)d_in[14];
  const float* b_up   = (const float*)d_in[15];

  float* out = (float*)d_out;
  float* flowout = out + (size_t)Bn * OCh * 512 * 512;                  // +33554432
  float* Wsout   = flowout + (size_t)Bn * 2 * Np;                       // +34078720

  // Workspace layout (256B aligned), ~305 MB total.
  char* ws = (char*)d_ws;
  size_t off = 0;
  auto alloc = [&](size_t bytes) -> char* {
    char* r = ws + off;
    off = (off + bytes + 255) & ~(size_t)255;
    return r;
  };
  // scratch: [0..23] image/entropy minmax; [32..287]/[288..543] branch1
  // final max/sum; [544..799]/[800..1055] branch2; then softmax partials.
  float*  scr    = (float*)alloc(16384);
  float*  pm1    = (float*)alloc(64 * 256 * 4);              // per-chunk partials
  float*  ps1    = (float*)alloc(64 * 256 * 4);
  float*  pm2    = (float*)alloc(64 * 256 * 4);
  float*  ps2    = (float*)alloc(64 * 256 * 4);
  half_t* pk_c11 = (half_t*)alloc(4 * 1 * 4 * 32 * 16 * 2);  // packed weights
  half_t* pk_ca1 = (half_t*)alloc(4 * 9 * 2 * 32 * 16 * 2);
  half_t* pk_ca2 = (half_t*)alloc(4 * 9 * 2 * 32 * 16 * 2);
  half_t* pk_fo  = (half_t*)alloc(4 * 9 * 2 * 32 * 16 * 2);
  half_t* pk_up  = (half_t*)alloc(2 * 1 * 2 * 32 * 16 * 2);
  float*  bins0  = (float*)alloc((size_t)Bn * 16 * Np * 4);  // 16.8 MB
  float*  bins1  = (float*)alloc((size_t)Bn * 16 * Np * 4);  // 16.8 MB
  float*  H1     = (float*)alloc((size_t)Bn * Np * 4);
  float*  H2     = (float*)alloc((size_t)Bn * Np * 4);
  float*  Wpre   = (float*)alloc((size_t)Bn * Np * 4);
  half_t* f1h    = (half_t*)alloc((size_t)Bn * 64 * Np * 2); // 33.5 MB each
  half_t* f2h    = (half_t*)alloc((size_t)Bn * 64 * Np * 2);
  half_t* A1h    = (half_t*)alloc((size_t)Bn * 64 * Np * 2);
  half_t* B2h    = (half_t*)alloc((size_t)Bn * 64 * Np * 2);
  half_t* A2h    = (half_t*)alloc((size_t)Bn * 64 * Np * 2);
  half_t* B1h    = (half_t*)alloc((size_t)Bn * 64 * Np * 2);
  half_t* fh     = (half_t*)alloc((size_t)Bn * 64 * Np * 2);
  half_t* fnh    = (half_t*)alloc((size_t)Bn * 64 * Np * 2);

  // --- prep: scratch + packed weights ---
  k_init<<<1, 64, 0, stream>>>(scr);
  k_pack<<<2, 256, 0, stream>>>(w_c11, pk_c11, 64, 128, 1);
  k_pack<<<9, 256, 0, stream>>>(w_ca1, pk_ca1, 64, 64, 9);
  k_pack<<<9, 256, 0, stream>>>(w_ca2, pk_ca2, 64, 64, 9);
  k_pack<<<9, 256, 0, stream>>>(w_fo,  pk_fo,  64, 64, 9);
  k_pack<<<1, 256, 0, stream>>>(w_up,  pk_up,  32, 64, 1);

  // --- entropy-difference weight path ---
  k_minmax2<<<dim3(16, Bn), 256, 0, stream>>>(img1, nullptr, 1.f, scr + 0, scr + 4);
  k_minmax2<<<dim3(16, Bn), 256, 0, stream>>>(img2, nullptr, 1.f, scr + 8, scr + 12);

  k_bins<<<1024, 256, 0, stream>>>(img1, scr + 0, scr + 4, bins0);
  k_poolh<<<16384, 256, 0, stream>>>(bins0, bins1);
  k_poolv_ent<<<1024, 256, 0, stream>>>(bins1, H1);

  k_bins<<<1024, 256, 0, stream>>>(img2, scr + 8, scr + 12, bins0);
  k_poolh<<<16384, 256, 0, stream>>>(bins0, bins1);
  k_poolv_ent<<<1024, 256, 0, stream>>>(bins1, H2);

  k_minmax2<<<dim3(16, Bn), 256, 0, stream>>>(H1, H2, 0.5f, scr + 16, scr + 20);
  k_weight<<<1024, 256, 0, stream>>>(H1, H2, scr + 16, scr + 20, Wpre);
  k_smooth<<<1024, 256, 0, stream>>>(Wpre, Wsout);

  // --- feature path (channels-last f16) ---
  k_tr<<<dim3(2048, Bn), 256, 0, stream>>>(f1, f1h);
  k_conv1x1_f2<<<8192, 256, 0, stream>>>(f2cat, pk_c11, b_c11, f2h);

  dim3 g3(8, 256, 4);
  k_conv3x3<<<g3, 256, 0, stream>>>(f1h, pk_ca1, b_ca1, A1h, nullptr);
  k_conv3x3<<<g3, 256, 0, stream>>>(f2h, pk_ca2, b_ca2, B2h, nullptr);
  k_conv3x3<<<g3, 256, 0, stream>>>(f2h, pk_ca1, b_ca1, A2h, nullptr);
  k_conv3x3<<<g3, 256, 0, stream>>>(f1h, pk_ca2, b_ca2, B1h, nullptr);

  // single-pass online softmax (partials + tree merge)
  dim3 gr(64, Bn);
  k_attpart<<<gr, 256, 0, stream>>>(A1h, B2h, Wsout, pm1, ps1);
  k_attpart<<<gr, 256, 0, stream>>>(A2h, B1h, Wsout, pm2, ps2);
  k_attfinal<<<1, 256, 0, stream>>>(pm1, ps1, scr + 32, scr + 288);
  k_attfinal<<<1, 256, 0, stream>>>(pm2, ps2, scr + 544, scr + 800);

  k_combine<<<65536, 256, 0, stream>>>(A1h, B2h, A2h, B1h, f1h, f2h, Wsout,
                                       scr + 32, scr + 288, scr + 544, scr + 800, fh);

  // f = f + relu(conv3x3(f, w_fo))
  k_conv3x3<<<g3, 256, 0, stream>>>(fh, pk_fo, b_fo, fnh, fh);

  // flow and upsampled features
  k_flow<<<1024, 256, 0, stream>>>(fnh, w_fl, b_fl, Wsout, flowout);
  k_upconv<<<dim3(8, 512, 4), 256, 0, stream>>>(fnh, pk_up, b_up, out);
}
